// UOTSolver_11897059410794
// MI455X (gfx1250) — compile-verified, matrix operand order checked by reading, not run on hardware
//
#include <hip/hip_runtime.h>
#include <math.h>

// ---------------------------------------------------------------------------
// UOT / Sinkhorn solver for MI455X (gfx1250, wave32, WMMA).
//   A = source_proto [512, 256]  f32
//   B = target_feat  [65536, 256] f32
// Outputs (flat f32, reference return order):
//   cost[512,65536] | cost_ext[513,65536] | plan[513,65536] | loss | dustbin |
//   class_scores[65536,512] | dustbin_scores[65536] | assignment[65536]
// ---------------------------------------------------------------------------

typedef __attribute__((ext_vector_type(16))) __bf16 v16bf;
typedef __attribute__((ext_vector_type(8)))  __bf16 v8bf;
typedef __attribute__((ext_vector_type(8)))  float  v8f;

#define KROWS 512
#define RTOT  513
#define NCOL  65536
#define DIM   256
#define NBINS 4096

// d_out offsets (float elements)
#define OFF_COST 0ul
#define OFF_CEXT 33554432ul     // 512*65536
#define OFF_PLAN 67174400ul     // + 513*65536
#define OFF_LOSS 100794368ul
#define OFF_DUST 100794369ul
#define OFF_CLS  100794370ul
#define OFF_DSC  134348802ul
#define OFF_ASN  134414338ul

static __device__ __forceinline__ float blk_reduce_sum(float x, float* sm) {
  const int t = threadIdx.x;
  sm[t] = x;
  __syncthreads();
  for (int s = 128; s > 0; s >>= 1) {
    if (t < s) sm[t] += sm[t + s];
    __syncthreads();
  }
  float r = sm[0];
  __syncthreads();
  return r;
}

// ---- init: hist/stats/u/v/loss slots (d_out & d_ws are poisoned 0xAA) -----
__global__ __launch_bounds__(256) void k_init(unsigned* stats, unsigned* hist,
                                              float* u, float* v,
                                              float* lossPart, float* out) {
  const int t = blockIdx.x * blockDim.x + threadIdx.x;  // 65536 threads
  if (t < NBINS) hist[t] = 0u;
  if (t < RTOT)  u[t] = 1.0f;
  if (t < 256)   lossPart[t] = 0.0f;
  v[t] = 1.0f;
  if (t == 0) {
    stats[0] = 0x7f800000u;  // +inf ; cost >= 0 so uint order == float order
    stats[1] = 0u;           // 0.0f
    out[OFF_LOSS] = 0.0f;
  }
}

// ---- fp32 -> bf16 conversion + exact fp32 squared row norms ---------------
__global__ __launch_bounds__(256) void k_prep(const float* __restrict__ src,
                                              unsigned short* __restrict__ dstBf,
                                              float* __restrict__ norms) {
  __shared__ float sm[256];
  const int r = blockIdx.x;
  const int d = threadIdx.x;
  const float x = src[(size_t)r * DIM + d];
  reinterpret_cast<__bf16*>(dstBf)[(size_t)r * DIM + d] = (__bf16)x;
  const float s = blk_reduce_sum(x * x, sm);
  if (d == 0) norms[r] = s;
}

// ---- WMMA GEMM: cost = sqrt(max(|a|^2 + |b|^2 - 2 A.B^T, 0)) -------------
static __device__ __forceinline__ v16bf load_frag(const __bf16* rowBase, int d0) {
  union { v16bf v; v8bf h[2]; } u;
  u.h[0] = *reinterpret_cast<const v8bf*>(rowBase + d0);
  u.h[1] = *reinterpret_cast<const v8bf*>(rowBase + d0 + 16);
  return u.v;
}

__global__ __launch_bounds__(256) void k_gemm(const unsigned short* __restrict__ Abf_,
                                              const unsigned short* __restrict__ Bbf_,
                                              const float* __restrict__ anorm,
                                              const float* __restrict__ bnorm,
                                              float* __restrict__ out) {
  const __bf16* Abf = reinterpret_cast<const __bf16*>(Abf_);
  const __bf16* Bbf = reinterpret_cast<const __bf16*>(Bbf_);
  const int lane  = threadIdx.x & 31;
  const int wave  = threadIdx.x >> 5;
  const int mBase = blockIdx.y * 128 + wave * 16;   // 8 waves stacked in M
  const int nBase = blockIdx.x * 64;                // 4 accumulators in N
  // 16-bit fragment K-offset pattern: lanes 0-15 take K{0-7,16-23}, 16-31 take K{8-15,24-31}
  const int lo = (lane & 16) ? 8 : 0;
  const int lr = lane & 15;
  const __bf16* aRow  = Abf + (size_t)(mBase + lr) * DIM;
  const __bf16* bRow0 = Bbf + (size_t)(nBase +  0 + lr) * DIM;
  const __bf16* bRow1 = Bbf + (size_t)(nBase + 16 + lr) * DIM;
  const __bf16* bRow2 = Bbf + (size_t)(nBase + 32 + lr) * DIM;
  const __bf16* bRow3 = Bbf + (size_t)(nBase + 48 + lr) * DIM;
  v8f c0 = {}, c1 = {}, c2 = {}, c3 = {};
#pragma unroll
  for (int kt = 0; kt < DIM / 32; ++kt) {
    const int d0 = kt * 32 + lo;
    const v16bf a = load_frag(aRow, d0);
    c0 = __builtin_amdgcn_wmma_f32_16x16x32_bf16(false, a, false, load_frag(bRow0, d0), (short)0, c0, false, false);
    c1 = __builtin_amdgcn_wmma_f32_16x16x32_bf16(false, a, false, load_frag(bRow1, d0), (short)0, c1, false, false);
    c2 = __builtin_amdgcn_wmma_f32_16x16x32_bf16(false, a, false, load_frag(bRow2, d0), (short)0, c2, false, false);
    c3 = __builtin_amdgcn_wmma_f32_16x16x32_bf16(false, a, false, load_frag(bRow3, d0), (short)0, c3, false, false);
  }
  // C/D layout: lanes 0-15 -> M = VGPRi, N = lane; lanes 16-31 -> M = VGPRi+8, N = lane-16
  const int mOff = (lane >> 4) << 3;
  v8f cc[4] = {c0, c1, c2, c3};
#pragma unroll
  for (int j = 0; j < 4; ++j) {
    const int n = nBase + j * 16 + lr;
    const float bn = bnorm[n];
#pragma unroll
    for (int i = 0; i < 8; ++i) {
      const int m = mBase + i + mOff;
      const float sq  = anorm[m] + bn - 2.0f * cc[j][i];
      const float cst = sqrtf(fmaxf(sq, 0.0f));
      const size_t idx = (size_t)m * NCOL + n;
      out[OFF_COST + idx] = cst;
      out[OFF_CEXT + idx] = cst;
    }
  }
}

// ---- quantile: min/max -> LDS histogram -> bin-interpolated 0.8 quantile --
__global__ __launch_bounds__(256) void k_minmax(const float* __restrict__ out,
                                                unsigned* __restrict__ stats) {
  __shared__ float smn[256], smx[256];
  float mn = 3.4e38f, mx = 0.0f;
  const size_t total = (size_t)KROWS * NCOL;
  for (size_t i = (size_t)blockIdx.x * 256 + threadIdx.x; i < total; i += (size_t)gridDim.x * 256) {
    const float x = out[OFF_COST + i];
    mn = fminf(mn, x);
    mx = fmaxf(mx, x);
  }
  const int t = threadIdx.x;
  smn[t] = mn; smx[t] = mx;
  __syncthreads();
  for (int s = 128; s > 0; s >>= 1) {
    if (t < s) { smn[t] = fminf(smn[t], smn[t + s]); smx[t] = fmaxf(smx[t], smx[t + s]); }
    __syncthreads();
  }
  if (t == 0) {
    atomicMin(&stats[0], __float_as_uint(smn[0]));
    atomicMax(&stats[1], __float_as_uint(smx[0]));
  }
}

__global__ __launch_bounds__(256) void k_hist(const float* __restrict__ out,
                                              const unsigned* __restrict__ stats,
                                              unsigned* __restrict__ hist) {
  __shared__ unsigned h[NBINS];  // 16 KB LDS
  for (int i = threadIdx.x; i < NBINS; i += 256) h[i] = 0u;
  __syncthreads();
  const float mn = __uint_as_float(stats[0]);
  const float mx = __uint_as_float(stats[1]);
  const float scale = (mx > mn) ? (float)NBINS / (mx - mn) : 0.0f;
  const size_t total = (size_t)KROWS * NCOL;
  for (size_t i = (size_t)blockIdx.x * 256 + threadIdx.x; i < total; i += (size_t)gridDim.x * 256) {
    int b = (int)((out[OFF_COST + i] - mn) * scale);
    b = min(max(b, 0), NBINS - 1);
    atomicAdd(&h[b], 1u);
  }
  __syncthreads();
  for (int i = threadIdx.x; i < NBINS; i += 256)
    if (h[i]) atomicAdd(&hist[i], h[i]);
}

__global__ void k_quantile(const unsigned* __restrict__ stats,
                           const unsigned* __restrict__ hist,
                           float* __restrict__ out) {
  const float mn = __uint_as_float(stats[0]);
  const float mx = __uint_as_float(stats[1]);
  const double total = (double)KROWS * (double)NCOL;
  const double rank = 0.8 * (total - 1.0);
  const float binw = (mx - mn) / (float)NBINS;
  double cum = 0.0;
  float q = mx;
  for (int b = 0; b < NBINS; ++b) {
    const double c = (double)hist[b];
    if (cum + c > rank) {
      const double fr = (c > 0.0) ? (rank - cum) / c : 0.0;
      q = mn + ((float)b + (float)fr) * binw;
      break;
    }
    cum += c;
  }
  out[OFF_DUST] = q;
}

// ---- cost_ext dustbin row + Gibbs kernel K = max(exp(-20*cost), 1e-8) -----
__global__ __launch_bounds__(256) void k_buildK(float* __restrict__ out,
                                                float* __restrict__ Kmat) {
  const float dust = out[OFF_DUST];
  const size_t total = (size_t)RTOT * NCOL;
  for (size_t i = (size_t)blockIdx.x * 256 + threadIdx.x; i < total; i += (size_t)gridDim.x * 256) {
    const int r = (int)(i >> 16);  // NCOL == 65536
    float x;
    if (r < KROWS) {
      x = out[OFF_CEXT + i];
    } else {
      x = dust;
      out[OFF_CEXT + i] = x;
    }
    Kmat[i] = fmaxf(__expf(-20.0f * x), 1e-8f);
  }
}

// ---- Sinkhorn half-iterations (K resident in 192MB L2) --------------------
__global__ __launch_bounds__(256) void k_row(const float* __restrict__ Kmat,
                                             const float* __restrict__ v,
                                             float* __restrict__ u) {
  __shared__ float sm[256];
  const int r = blockIdx.x;
  const float* Kr = Kmat + (size_t)r * NCOL;
  float acc = 0.0f;
  for (int n = threadIdx.x; n < NCOL; n += 256) acc += Kr[n] * v[n];
  const float s = blk_reduce_sum(acc, sm);
  if (threadIdx.x == 0) u[r] = (1.0f / (float)RTOT) / fmaxf(s, 1e-8f);  // tau_s == 1
}

__global__ __launch_bounds__(256) void k_col(const float* __restrict__ Kmat,
                                             const float* __restrict__ u,
                                             float* __restrict__ v) {
  __shared__ float su[RTOT];
  for (int i = threadIdx.x; i < RTOT; i += 256) su[i] = u[i];
  __syncthreads();
  const int n = blockIdx.x * 256 + threadIdx.x;  // coalesced over columns
  float acc = 0.0f;
  for (int r = 0; r < RTOT; ++r) acc += Kmat[(size_t)r * NCOL + n] * su[r];
  v[n] = (1.0f / (float)NCOL) / fmaxf(acc, 1e-8f);  // tau_t == 1
}

// ---- plan, loss partials, dustbin scores, argmax, 1/class-sum -------------
__global__ __launch_bounds__(256) void k_final(const float* __restrict__ Kmat,
                                               const float* __restrict__ u,
                                               const float* __restrict__ v,
                                               float* __restrict__ out,
                                               float* __restrict__ lossPart,
                                               float* __restrict__ invCls) {
  __shared__ float su[RTOT];
  __shared__ float red[256];
  for (int i = threadIdx.x; i < RTOT; i += 256) su[i] = u[i];
  __syncthreads();
  const int n = blockIdx.x * 256 + threadIdx.x;
  const float vn = v[n];
  float colsum = 0.0f, clssum = 0.0f, lacc = 0.0f, best = -1.0f, pLast = 0.0f;
  int bi = 0;
  for (int r = 0; r < RTOT; ++r) {
    const size_t idx = (size_t)r * NCOL + n;
    const float p = su[r] * Kmat[idx] * vn;
    out[OFF_PLAN + idx] = p;
    colsum += p;
    lacc += p * out[OFF_CEXT + idx];
    if (r < KROWS) {
      clssum += p;
      if (p > best) { best = p; bi = r; }  // strict > keeps first max (argmax semantics)
    } else {
      pLast = p;
    }
  }
  out[OFF_DSC + n] = pLast / fmaxf(colsum, 1e-8f);
  out[OFF_ASN + n] = (float)bi;
  invCls[n] = 1.0f / fmaxf(clssum, 1e-8f);
  const float ls = blk_reduce_sum(lacc, red);
  if (threadIdx.x == 0) lossPart[blockIdx.x] = ls;
}

// ---- class_scores = (plan[:512].T) * invCls : LDS-tiled 32x32 transpose ---
// Reads plan row-major (coalesced), writes [n,512] with consecutive lanes on
// consecutive r (coalesced) — avoids the 2KB-stride scatter of a naive loop.
__global__ __launch_bounds__(256) void k_cls(const float* __restrict__ out_ro,
                                             const float* __restrict__ invCls,
                                             float* __restrict__ out) {
  __shared__ float tile[32][33];  // +1 pad: no LDS bank conflicts
  const int nBase = blockIdx.x * 32;
  const int rBase = blockIdx.y * 32;   // rBase+31 <= 511: dustbin row excluded
  const int t  = threadIdx.x;
  const int c0 = t & 31;        // inner (coalesced) index
  const int g  = t >> 5;        // 0..7
#pragma unroll
  for (int i = 0; i < 4; ++i) {
    const int rl = g + i * 8;
    tile[rl][c0] = out_ro[OFF_PLAN + (size_t)(rBase + rl) * NCOL + (nBase + c0)];
  }
  __syncthreads();
#pragma unroll
  for (int i = 0; i < 4; ++i) {
    const int nl = g + i * 8;
    out[OFF_CLS + (size_t)(nBase + nl) * KROWS + (rBase + c0)] =
        tile[c0][nl] * invCls[nBase + nl];
  }
}

__global__ void k_loss(const float* __restrict__ lossPart, float* __restrict__ out) {
  double s = 0.0;
  for (int i = 0; i < 256; ++i) s += (double)lossPart[i];  // deterministic order
  out[OFF_LOSS] = (float)s;
}

// ---------------------------------------------------------------------------
extern "C" void kernel_launch(void* const* d_in, const int* in_sizes, int n_in,
                              void* d_out, int out_size, void* d_ws, size_t ws_size,
                              hipStream_t stream) {
  (void)in_sizes; (void)n_in; (void)out_size; (void)ws_size;
  const float* A = (const float*)d_in[0];   // [512, 256]
  const float* B = (const float*)d_in[1];   // [65536, 256]
  float* out = (float*)d_out;

  // Workspace layout (256-B aligned slabs, ~161 MB total)
  char* w = (char*)d_ws;
  unsigned short* Abf = (unsigned short*)(w + 0);          //   512*256 bf16
  unsigned short* Bbf = (unsigned short*)(w + 262144);     // 65536*256 bf16
  float* anorm    = (float*)(w + 33816576);                // 512 f32
  float* bnorm    = (float*)(w + 33818624);                // 65536 f32
  float* u        = (float*)(w + 34080768);                // 513 f32
  float* v        = (float*)(w + 34084864);                // 65536 f32
  unsigned* stats = (unsigned*)(w + 34347008);             // min/max bits
  unsigned* hist  = (unsigned*)(w + 34347264);             // 4096 u32
  float* lossPart = (float*)(w + 34363648);                // 256 f32
  float* invCls   = (float*)(w + 34364672);                // 65536 f32
  float* Kmat     = (float*)(w + 34626816);                // 513*65536 f32 (134.5 MB, fits L2)

  k_init<<<256, 256, 0, stream>>>(stats, hist, u, v, lossPart, out);
  k_prep<<<KROWS, 256, 0, stream>>>(A, Abf, anorm);
  k_prep<<<NCOL,  256, 0, stream>>>(B, Bbf, bnorm);
  k_gemm<<<dim3(NCOL / 64, KROWS / 128), 256, 0, stream>>>(Abf, Bbf, anorm, bnorm, out);
  k_minmax<<<1024, 256, 0, stream>>>(out, stats);
  k_hist<<<1024, 256, 0, stream>>>(out, stats, hist);
  k_quantile<<<1, 1, 0, stream>>>(stats, hist, out);
  k_buildK<<<4096, 256, 0, stream>>>(out, Kmat);
  for (int it = 0; it < 30; ++it) {
    k_row<<<RTOT, 256, 0, stream>>>(Kmat, v, u);
    k_col<<<NCOL / 256, 256, 0, stream>>>(Kmat, u, v);
  }
  k_final<<<NCOL / 256, 256, 0, stream>>>(Kmat, u, v, out, lossPart, invCls);
  k_cls<<<dim3(NCOL / 32, KROWS / 32), 256, 0, stream>>>(out, invCls, out);
  k_loss<<<1, 1, 0, stream>>>(lossPart, out);
}
